// RGDTLayer_73675868996339
// MI455X (gfx1250) — compile-verified, matrix operand order checked by reading, not run on hardware
//
#include <hip/hip_runtime.h>
#include <hip/hip_bf16.h>
#include <math.h>

// ---------------- sizes (fixed by the reference) ----------------
#define NN   50000
#define EE   800000
#define RR   32
#define HH   8
#define DD   16
#define DK   128      // H*D == DIN
#define FFW  512      // 4*H*D

typedef __attribute__((ext_vector_type(16))) __bf16 v16bf;
typedef __attribute__((ext_vector_type(8)))  float  v8f;

union FragAB { v16bf v; unsigned int u[8]; };
union FragD  { v8f  v; float f[8]; };

#define DEVFN static __device__ __forceinline__

// CDNA5 async memory->LDS path (ASYNCcnt), guarded so compile never breaks.
#if defined(__has_builtin)
#if __has_builtin(__builtin_amdgcn_global_load_async_to_lds_b128) && \
    __has_builtin(__builtin_amdgcn_s_wait_asynccnt)
#define HAVE_ASYNC_LDS 1
#endif
#endif
#ifndef HAVE_ASYNC_LDS
#define HAVE_ASYNC_LDS 0
#endif

#if HAVE_ASYNC_LDS
typedef int v4i_vs __attribute__((vector_size(4 * sizeof(int))));
#define GLOBAL_AS(p) ((__attribute__((address_space(1))) v4i_vs*)(p))
#define LDS_AS(p)    ((__attribute__((address_space(3))) v4i_vs*)(p))
#endif

DEVFN unsigned short f32_to_bf16(float x) {
  unsigned int u = __float_as_uint(x);
  u += 0x7FFFu + ((u >> 16) & 1u);           // round-to-nearest-even
  return (unsigned short)(u >> 16);
}

// order-preserving float<->uint encoding for atomic segment-max
DEVFN unsigned int enc_ord(float f) {
  unsigned int u = __float_as_uint(f);
  return (u & 0x80000000u) ? ~u : (u | 0x80000000u);
}
DEVFN float dec_ord(unsigned int k) {
  unsigned int u = (k & 0x80000000u) ? (k & 0x7FFFFFFFu) : ~k;
  return __uint_as_float(u);
}

// ---- WMMA bf16 16x16x32 fragment helpers (CDNA5 wave32 layout) ----
// A 16x32: lane m=lane&15, half=lane>>4; u[i] holds K-pair
//   k = k0 + 8*half + ((i&4)?16:0) + 2*(i&3)
// B 32x16 uses the same mapping with lane indexing N, so we store weights
// TRANSPOSED ([col][k], K contiguous) and reuse the same loader.
DEVFN FragAB load_frag_rm(const unsigned short* base, int row0, int ld, int k0, int lane) {
  int m = lane & 15, half = lane >> 4;
  const unsigned short* p = base + (size_t)(row0 + m) * ld + k0 + 8 * half;
  FragAB f;
#pragma unroll
  for (int i = 0; i < 8; ++i) {
    int off = ((i & 4) << 2) + 2 * (i & 3);  // 0,2,4,6,16,18,20,22
    f.u[i] = *(const unsigned int*)(p + off);
  }
  return f;
}

DEVFN v8f wmma_bf16(const FragAB& a, const FragAB& b, v8f c) {
  return __builtin_amdgcn_wmma_f32_16x16x32_bf16(
      /*neg_a=*/false, a.v, /*neg_b=*/false, b.v,
      /*c_mod=*/(short)0, c, /*reuse_a=*/false, /*reuse_b=*/false);
}

DEVFN void store_tile(float* C, int row0, int col0, int ld, const FragD& d, int lane) {
  int n = lane & 15, half = lane >> 4;
#pragma unroll
  for (int r = 0; r < 8; ++r)
    C[(size_t)(row0 + r + 8 * half) * ld + col0 + n] = d.f[r];
}

// ---------------- small utility kernels ----------------
__global__ void fill_kernel(float* p, float v, int n) {
  int i = blockIdx.x * 256 + threadIdx.x;
  if (i < n) p[i] = v;
}

// Wt[c*rows + r] = bf16(W[r*cols + c])   (rows = K dim)
__global__ void transpose_bf16_kernel(const float* __restrict__ W,
                                      unsigned short* __restrict__ Wt,
                                      int rows, int cols) {
  int i = blockIdx.x * 256 + threadIdx.x;
  if (i >= rows * cols) return;
  int r = i / cols, c = i - r * cols;
  Wt[(size_t)c * rows + r] = f32_to_bf16(W[i]);
}

// row-wise LayerNorm (128 wide) -> bf16
__global__ __launch_bounds__(128) void ln_bf16_kernel(const float* __restrict__ X,
    const float* __restrict__ g, const float* __restrict__ b,
    unsigned short* __restrict__ out, int nrows) {
  int row = blockIdx.x;
  if (row >= nrows) return;
  int t = threadIdx.x;
  float v = X[(size_t)row * DK + t];
  __shared__ float red[4];
  float s = v;
#pragma unroll
  for (int o = 16; o > 0; o >>= 1) s += __shfl_down(s, o, 32);
  if ((t & 31) == 0) red[t >> 5] = s;
  __syncthreads();
  float mean = (red[0] + red[1] + red[2] + red[3]) * (1.0f / 128.0f);
  __syncthreads();
  float dv = v - mean;
  s = dv * dv;
#pragma unroll
  for (int o = 16; o > 0; o >>= 1) s += __shfl_down(s, o, 32);
  if ((t & 31) == 0) red[t >> 5] = s;
  __syncthreads();
  float var = (red[0] + red[1] + red[2] + red[3]) * (1.0f / 128.0f);
  float y = dv * rsqrtf(var + 1e-5f) * g[t] + b[t];
  out[(size_t)row * DK + t] = f32_to_bf16(y);
}

// rst = feat + ent ; h = bf16(LN(rst))
__global__ __launch_bounds__(128) void residual_ln_kernel(const float* __restrict__ feat,
    const float* __restrict__ ent, const float* __restrict__ g, const float* __restrict__ b,
    float* __restrict__ rst, unsigned short* __restrict__ hbf, int nrows) {
  int row = blockIdx.x;
  if (row >= nrows) return;
  int t = threadIdx.x;
  size_t gi = (size_t)row * DK + t;
  float v = feat[gi] + ent[gi];
  rst[gi] = v;
  __shared__ float red[4];
  float s = v;
#pragma unroll
  for (int o = 16; o > 0; o >>= 1) s += __shfl_down(s, o, 32);
  if ((t & 31) == 0) red[t >> 5] = s;
  __syncthreads();
  float mean = (red[0] + red[1] + red[2] + red[3]) * (1.0f / 128.0f);
  __syncthreads();
  float dv = v - mean;
  s = dv * dv;
#pragma unroll
  for (int o = 16; o > 0; o >>= 1) s += __shfl_down(s, o, 32);
  if ((t & 31) == 0) red[t >> 5] = s;
  __syncthreads();
  float var = (red[0] + red[1] + red[2] + red[3]) * (1.0f / 128.0f);
  hbf[gi] = f32_to_bf16(dv * rsqrtf(var + 1e-5f) * g[t] + b[t]);
}

// C[M,128] = Abf16[M,128] @ W (W given transposed bf16 [128][128], staged in LDS)
__global__ __launch_bounds__(256) void gemm128_kernel(const unsigned short* __restrict__ A,
    const unsigned short* __restrict__ Wt, float* __restrict__ C, int M) {
  __shared__ unsigned short sW[DK * DK];               // 32 KB
  int tid = threadIdx.x;
#if HAVE_ASYNC_LDS
  // memory -> LDS DMA, 16B per lane-issue, 2048 chunks total
  for (int i = tid; i < DK * DK / 8; i += 256)
    __builtin_amdgcn_global_load_async_to_lds_b128(
        GLOBAL_AS(Wt + i * 8), LDS_AS(sW + i * 8), 0, 0);
  __builtin_amdgcn_s_wait_asynccnt(0);
#else
  for (int i = tid; i < DK * DK / 2; i += 256)
    ((unsigned int*)sW)[i] = ((const unsigned int*)Wt)[i];
#endif
  __syncthreads();
  int wave = tid >> 5, lane = tid & 31;
  int row0 = blockIdx.x * 128 + wave * 16;
  if (row0 >= M) return;                               // uniform per wave
  v8f z = {0.f, 0.f, 0.f, 0.f, 0.f, 0.f, 0.f, 0.f};
  FragD acc[8];
#pragma unroll
  for (int n = 0; n < 8; ++n) acc[n].v = z;
  for (int ks = 0; ks < 4; ++ks) {
    int k0 = ks * 32;
    FragAB a = load_frag_rm(A, row0, DK, k0, lane);
#pragma unroll
    for (int n = 0; n < 8; ++n) {
      FragAB b = load_frag_rm(sW, n * 16, DK, k0, lane);
      acc[n].v = wmma_bf16(a, b, acc[n].v);
    }
  }
#pragma unroll
  for (int n = 0; n < 8; ++n) store_tile(C, row0, n * 16, DK, acc[n], lane);
}

// er[r,h] = sum_d feat_rel[r,h,d] * attn_r[h,d]
__global__ void er_kernel(const float* __restrict__ frel, const float* __restrict__ attn,
                          float* __restrict__ er) {
  int i = blockIdx.x * 256 + threadIdx.x;
  if (i >= RR * HH) return;
  int r = i >> 3, h = i & 7;
  float acc = 0.f;
#pragma unroll
  for (int d = 0; d < DD; ++d) acc += frel[r * DK + h * DD + d] * attn[h * DD + d];
  er[i] = acc;
}

__global__ void deg_kernel(const int* __restrict__ dst, float* __restrict__ deg, int E) {
  int i = blockIdx.x * 256 + threadIdx.x;
  if (i < E) atomicAdd(&deg[dst[i]], 1.0f);
}

// e[e,h] = dot(fh[src],ft[dst]) * er[rid,h] * log(deg[dst]) / sqrt(D)
__global__ void edge_score_kernel(const float* __restrict__ fh, const float* __restrict__ ft,
    const float* __restrict__ er, const int* __restrict__ src, const int* __restrict__ dst,
    const int* __restrict__ rid, const float* __restrict__ deg, float* __restrict__ e, int EH) {
  int idx = blockIdx.x * 256 + threadIdx.x;
  if (idx >= EH) return;
  int h = idx & 7, ed = idx >> 3;
  int s = src[ed], d = dst[ed], r = rid[ed];
  const float4* ph = (const float4*)(fh + (size_t)s * DK + h * DD);
  const float4* pt = (const float4*)(ft + (size_t)d * DK + h * DD);
  float acc = 0.f;
#pragma unroll
  for (int i = 0; i < 4; ++i) {
    float4 a = ph[i], b = pt[i];
    acc += a.x * b.x + a.y * b.y + a.z * b.z + a.w * b.w;
  }
  float li = logf(fmaxf(deg[d], 1.0f));
  e[idx] = acc * er[r * HH + h] * li * 0.25f;          // 1/sqrt(16)
}

__global__ void smax_max_kernel(const float* __restrict__ e, const int* __restrict__ dst,
                                unsigned int* __restrict__ m, int EH) {
  int idx = blockIdx.x * 256 + threadIdx.x;
  if (idx >= EH) return;
  int h = idx & 7, ed = idx >> 3;
  atomicMax(&m[dst[ed] * HH + h], enc_ord(e[idx]));
}

__global__ void smax_exp_kernel(float* __restrict__ e, const int* __restrict__ dst,
                                const unsigned int* __restrict__ m, float* __restrict__ s, int EH) {
  int idx = blockIdx.x * 256 + threadIdx.x;
  if (idx >= EH) return;
  int h = idx & 7, ed = idx >> 3;
  int seg = dst[ed] * HH + h;
  float ex = expf(e[idx] - dec_ord(m[seg]));
  e[idx] = ex;
  atomicAdd(&s[seg], ex);
}

__global__ void smax_div_kernel(float* __restrict__ e, const int* __restrict__ dst,
                                const float* __restrict__ s, int EH) {
  int idx = blockIdx.x * 256 + threadIdx.x;
  if (idx >= EH) return;
  int h = idx & 7, ed = idx >> 3;
  e[idx] = e[idx] / s[dst[ed] * HH + h];
}

// agg[dst] += feat[src] * a[e,h]  (16 floats per (edge,head))
__global__ void scatter_kernel(const float* __restrict__ feat, const float* __restrict__ a,
    const int* __restrict__ src, const int* __restrict__ dst, float* __restrict__ agg, int EH) {
  int idx = blockIdx.x * 256 + threadIdx.x;
  if (idx >= EH) return;
  int h = idx & 7, ed = idx >> 3;
  float w = a[idx];
  const float* pf = feat + (size_t)src[ed] * DK + h * DD;
  float* pa = agg + (size_t)dst[ed] * DK + h * DD;
#pragma unroll
  for (int i = 0; i < DD; ++i) atomicAdd(pa + i, pf[i] * w);
}

// feat = 0.9*agg + 0.1*fe ; agg = 0 (ready for next hop)
__global__ void combine_clear_kernel(float* __restrict__ agg, const float* __restrict__ fe,
                                     float* __restrict__ feat, int n) {
  int i = blockIdx.x * 256 + threadIdx.x;
  if (i < n) {
    feat[i] = 0.9f * agg[i] + 0.1f * fe[i];
    agg[i] = 0.f;
  }
}

// Fused FFN per 32-row tile: out = relu(h@W1+b1)@W2 + b2 + rst
__global__ __launch_bounds__(256) void ffn_kernel(const unsigned short* __restrict__ Hb,
    const unsigned short* __restrict__ Wt1,   // [512][128] bf16
    const float* __restrict__ b1,
    const unsigned short* __restrict__ Wt2,   // [128][512] bf16
    const float* __restrict__ b2,
    const float* __restrict__ rst, float* __restrict__ out, int Ntot) {
  __shared__ unsigned short sH[32 * DK];      // 8 KB
  __shared__ unsigned short sMid[32 * FFW];   // 32 KB
  int tid = threadIdx.x;
  int row0 = blockIdx.x * 32;
  // stage h tile; rows past N get duplicated data (results discarded)
#if HAVE_ASYNC_LDS
  for (int i = tid; i < 32 * DK / 8; i += 256) {        // 512 chunks of 16B
    int g = row0 * (DK / 8) + i;                        // global chunk index
    int gc = (g < NN * (DK / 8)) ? g : (NN * (DK / 8) - 1);
    __builtin_amdgcn_global_load_async_to_lds_b128(
        GLOBAL_AS(Hb + (size_t)gc * 8), LDS_AS(sH + i * 8), 0, 0);
  }
  __builtin_amdgcn_s_wait_asynccnt(0);
#else
  for (int i = tid; i < 32 * DK / 2; i += 256) {
    int gi = row0 * (DK / 2) + i;
    ((unsigned int*)sH)[i] = (gi < Ntot * (DK / 2)) ? ((const unsigned int*)Hb)[gi] : 0u;
  }
#endif
  __syncthreads();
  int wave = tid >> 5, lane = tid & 31;
  int n = lane & 15, half = lane >> 4;
  v8f z = {0.f, 0.f, 0.f, 0.f, 0.f, 0.f, 0.f, 0.f};
  // GEMM1: mid[32,512]; wave w -> ntiles {w, w+8, w+16, w+24}, both mtiles
  FragD acc[2][4];
#pragma unroll
  for (int mt = 0; mt < 2; ++mt)
#pragma unroll
    for (int j = 0; j < 4; ++j) acc[mt][j].v = z;
  for (int ks = 0; ks < 4; ++ks) {
    int k0 = ks * 32;
    FragAB a0 = load_frag_rm(sH, 0, DK, k0, lane);
    FragAB a1 = load_frag_rm(sH, 16, DK, k0, lane);
#pragma unroll
    for (int j = 0; j < 4; ++j) {
      int nt = wave + j * 8;
      FragAB b = load_frag_rm(Wt1, nt * 16, DK, k0, lane);
      acc[0][j].v = wmma_bf16(a0, b, acc[0][j].v);
      acc[1][j].v = wmma_bf16(a1, b, acc[1][j].v);
    }
  }
  // bias + relu -> LDS (bf16)
#pragma unroll
  for (int mt = 0; mt < 2; ++mt)
#pragma unroll
    for (int j = 0; j < 4; ++j) {
      int col = (wave + j * 8) * 16 + n;
      float bb = b1[col];
#pragma unroll
      for (int r = 0; r < 8; ++r) {
        int rowm = mt * 16 + r + 8 * half;
        float v = acc[mt][j].f[r] + bb;
        sMid[rowm * FFW + col] = f32_to_bf16(fmaxf(v, 0.f));
      }
    }
  __syncthreads();
  // GEMM2: out[32,128]; wave w -> ntile w, both mtiles; K=512
  FragD o0, o1;
  o0.v = z; o1.v = z;
  for (int ks = 0; ks < 16; ++ks) {
    int k0 = ks * 32;
    FragAB a0 = load_frag_rm(sMid, 0, FFW, k0, lane);
    FragAB a1 = load_frag_rm(sMid, 16, FFW, k0, lane);
    FragAB b = load_frag_rm(Wt2, wave * 16, FFW, k0, lane);
    o0.v = wmma_bf16(a0, b, o0.v);
    o1.v = wmma_bf16(a1, b, o1.v);
  }
  int col = wave * 16 + n;
  float bb = b2[col];
#pragma unroll
  for (int mt = 0; mt < 2; ++mt) {
    const FragD& o = mt ? o1 : o0;
#pragma unroll
    for (int r = 0; r < 8; ++r) {
      int row = row0 + mt * 16 + r + 8 * half;
      if (row < Ntot) {
        size_t gi = (size_t)row * DK + col;
        out[gi] = o.f[r] + bb + rst[gi];
      }
    }
  }
}

// ---------------- host side ----------------
extern "C" void kernel_launch(void* const* d_in, const int* in_sizes, int n_in,
                              void* d_out, int out_size, void* d_ws, size_t ws_size,
                              hipStream_t stream) {
  const float* ent      = (const float*)d_in[0];
  const float* rel      = (const float*)d_in[1];
  const int*   src      = (const int*)d_in[2];
  const int*   dst      = (const int*)d_in[3];
  const int*   rid      = (const int*)d_in[4];
  const float* W_head   = (const float*)d_in[5];
  const float* W_tail   = (const float*)d_in[6];
  const float* W_ent    = (const float*)d_in[7];
  const float* W_rel    = (const float*)d_in[8];
  const float* attn_r   = (const float*)d_in[9];
  const float* ln_ent_g = (const float*)d_in[10];
  const float* ln_ent_b = (const float*)d_in[11];
  const float* ln_rel_g = (const float*)d_in[12];
  const float* ln_rel_b = (const float*)d_in[13];
  const float* ff_ln_g  = (const float*)d_in[14];
  const float* ff_ln_b  = (const float*)d_in[15];
  const float* W1       = (const float*)d_in[16];
  const float* b1       = (const float*)d_in[17];
  const float* W2       = (const float*)d_in[18];
  const float* b2       = (const float*)d_in[19];
  float* out = (float*)d_out;

  char* ws = (char*)d_ws;
  size_t off = 0;
  auto alloc = [&](size_t bytes) {
    void* p = ws + off;
    off = (off + bytes + 255) & ~(size_t)255;
    return p;
  };
  unsigned short* WT_HEAD = (unsigned short*)alloc(DK * DK * 2);
  unsigned short* WT_TAIL = (unsigned short*)alloc(DK * DK * 2);
  unsigned short* WT_ENT  = (unsigned short*)alloc(DK * DK * 2);
  unsigned short* WT_REL  = (unsigned short*)alloc(DK * DK * 2);
  unsigned short* WT1     = (unsigned short*)alloc(DK * FFW * 2);
  unsigned short* WT2     = (unsigned short*)alloc(FFW * DK * 2);
  unsigned short* XLN     = (unsigned short*)alloc((size_t)NN * DK * 2);
  unsigned short* RLN     = (unsigned short*)alloc((size_t)RR * DK * 2);
  float* FH   = (float*)alloc((size_t)NN * DK * 4);
  float* FT   = (float*)alloc((size_t)NN * DK * 4);
  float* FE   = (float*)alloc((size_t)NN * DK * 4);
  float* FREL = (float*)alloc((size_t)RR * DK * 4);
  float* ER   = (float*)alloc(RR * HH * 4);
  float* DEG  = (float*)alloc((size_t)NN * 4);
  float* EBUF = (float*)alloc((size_t)EE * HH * 4);
  unsigned int* MBUF = (unsigned int*)alloc((size_t)NN * HH * 4);
  float* SBUF = (float*)alloc((size_t)NN * HH * 4);
  float* FEAT = (float*)alloc((size_t)NN * DK * 4);
  float* AGG  = (float*)alloc((size_t)NN * DK * 4);
  float* RST  = (float*)alloc((size_t)NN * DK * 4);
  unsigned short* HBF = (unsigned short*)alloc((size_t)NN * DK * 2);

  auto cdiv = [](int a, int b) { return (a + b - 1) / b; };
  const int EH = EE * HH;
  const int NF = NN * DK;

  // weight convert + transpose (bf16, K-contiguous)
  transpose_bf16_kernel<<<cdiv(DK * DK, 256), 256, 0, stream>>>(W_head, WT_HEAD, DK, DK);
  transpose_bf16_kernel<<<cdiv(DK * DK, 256), 256, 0, stream>>>(W_tail, WT_TAIL, DK, DK);
  transpose_bf16_kernel<<<cdiv(DK * DK, 256), 256, 0, stream>>>(W_ent,  WT_ENT,  DK, DK);
  transpose_bf16_kernel<<<cdiv(DK * DK, 256), 256, 0, stream>>>(W_rel,  WT_REL,  DK, DK);
  transpose_bf16_kernel<<<cdiv(DK * FFW, 256), 256, 0, stream>>>(W1, WT1, DK, FFW);
  transpose_bf16_kernel<<<cdiv(FFW * DK, 256), 256, 0, stream>>>(W2, WT2, FFW, DK);

  // layer norms -> bf16 activations
  ln_bf16_kernel<<<NN, 128, 0, stream>>>(ent, ln_ent_g, ln_ent_b, XLN, NN);
  ln_bf16_kernel<<<RR, 128, 0, stream>>>(rel, ln_rel_g, ln_rel_b, RLN, RR);

  // WMMA GEMMs: feat_head/tail/ent, feat_rel
  gemm128_kernel<<<cdiv(NN, 128), 256, 0, stream>>>(XLN, WT_HEAD, FH, NN);
  gemm128_kernel<<<cdiv(NN, 128), 256, 0, stream>>>(XLN, WT_TAIL, FT, NN);
  gemm128_kernel<<<cdiv(NN, 128), 256, 0, stream>>>(XLN, WT_ENT,  FE, NN);
  gemm128_kernel<<<1, 256, 0, stream>>>(RLN, WT_REL, FREL, RR);

  er_kernel<<<1, 256, 0, stream>>>(FREL, attn_r, ER);

  // degrees + edge scores
  fill_kernel<<<cdiv(NN, 256), 256, 0, stream>>>(DEG, 0.f, NN);
  fill_kernel<<<cdiv(NN * HH, 256), 256, 0, stream>>>((float*)MBUF, 0.f, NN * HH);
  fill_kernel<<<cdiv(NN * HH, 256), 256, 0, stream>>>(SBUF, 0.f, NN * HH);
  deg_kernel<<<cdiv(EE, 256), 256, 0, stream>>>(dst, DEG, EE);
  edge_score_kernel<<<cdiv(EH, 256), 256, 0, stream>>>(FH, FT, ER, src, dst, rid, DEG, EBUF, EH);

  // segment softmax over dst (EBUF becomes attention `a`)
  smax_max_kernel<<<cdiv(EH, 256), 256, 0, stream>>>(EBUF, dst, MBUF, EH);
  smax_exp_kernel<<<cdiv(EH, 256), 256, 0, stream>>>(EBUF, dst, MBUF, SBUF, EH);
  smax_div_kernel<<<cdiv(EH, 256), 256, 0, stream>>>(EBUF, dst, SBUF, EH);

  // PPR diffusion (5 hops); combine re-zeroes AGG for the next hop
  fill_kernel<<<cdiv(NF, 256), 256, 0, stream>>>(AGG, 0.f, NF);
  for (int hop = 0; hop < 5; ++hop) {
    const float* fsrc = (hop == 0) ? FE : FEAT;
    scatter_kernel<<<cdiv(EH, 256), 256, 0, stream>>>(fsrc, EBUF, src, dst, AGG, EH);
    combine_clear_kernel<<<cdiv(NF, 256), 256, 0, stream>>>(AGG, FE, FEAT, NF);
  }

  // residual + LN + fused WMMA FFN
  residual_ln_kernel<<<NN, 128, 0, stream>>>(FEAT, ent, ff_ln_g, ff_ln_b, RST, HBF, NN);
  ffn_kernel<<<cdiv(NN, 32), 256, 0, stream>>>(HBF, WT1, b1, WT2, b2, RST, out, NN);

  (void)in_sizes; (void)n_in; (void)out_size; (void)ws_size;
}
